// KroneckerDecompAttention_59536836657210
// MI455X (gfx1250) — matrix operand order repaired and checked by reading, not compiled, not run on hardware
//
#include <hip/hip_runtime.h>
#include <math.h>

typedef __bf16 bf16;
typedef __attribute__((ext_vector_type(16))) __bf16 bf16x16;
typedef __attribute__((ext_vector_type(8)))  __bf16 bf16x8;
typedef __attribute__((ext_vector_type(8)))  float  v8f;
typedef int v4i __attribute__((vector_size(16)));
typedef __attribute__((address_space(1))) v4i gv4i;   // global int4
typedef __attribute__((address_space(3))) v4i lv4i;   // LDS int4

#define BH    32
#define NSEQ  8192
#define DIM   64
#define GRP   8
#define PLEN  1024
#define MSMP  34            // max(1, int(1024/30))
#define NSUB  (GRP * MSMP)  // 272
#define EPS_F 1.1920929e-07f
#define LOG_G 2.0794415416798357f  // ln(8)
#define NEG_BIG (-1e30f)

constexpr int KT_S = 80;  // K tile LDS row stride (bf16), 160B rows (16B aligned)
constexpr int VT_S = 40;  // V^T tile row stride, 80B rows
constexpr int PT_S = 40;  // P tile row stride

// CDNA5 async global->LDS path (ASYNCcnt-tracked, double buffered). Guarded so
// the file still compiles on toolchains without the builtins.
#if defined(__AMDGCN__) && __has_builtin(__builtin_amdgcn_global_load_async_to_lds_b128) && \
    __has_builtin(__builtin_amdgcn_s_wait_asynccnt)
#define HAS_ASYNC 1
#else
#define HAS_ASYNC 0
#endif

// ---------------- fragment loaders ----------------
// A-matrix 16x32 bf16 (ISA 7.12.2): lanes 0-15 row M=lane, K in [0..7]∪[16..23];
// lanes 16-31 same rows, K in [8..15]∪[24..31]  -> two 8-elem chunks.
__device__ __forceinline__ bf16x16 load_fragA(const bf16* base, int row, int stride,
                                              int kbase, int lane) {
  const int h = (lane & 16) ? 8 : 0;
  const bf16* p = base + row * stride + kbase + h;
  bf16x8 lo = *(const bf16x8*)(p);
  bf16x8 hi = *(const bf16x8*)(p + 16);
  bf16x16 f;
#pragma unroll
  for (int i = 0; i < 8; ++i) { f[i] = lo[i]; f[i + 8] = hi[i]; }
  return f;
}
// B-matrix 32x16 bf16: lanes 0-15 col N=lane with K in [0..7]∪[16..23], lanes 16-31
// K in [8..15]∪[24..31] (matches A). "row" here indexes the row-major source holding B^T.
__device__ __forceinline__ bf16x16 load_fragB(const bf16* base, int row, int stride,
                                              int kbase, int lane) {
  const int h = (lane & 16) ? 8 : 0;
  const bf16* p = base + row * stride + kbase + h;
  bf16x8 lo = *(const bf16x8*)(p);
  bf16x8 hi = *(const bf16x8*)(p + 16);
  bf16x16 f;
#pragma unroll
  for (int i = 0; i < 8; ++i) { f[i] = lo[i]; f[i + 8] = hi[i]; }
  return f;
}

__device__ __forceinline__ float red_max16(float v) {
#pragma unroll
  for (int m = 1; m < 16; m <<= 1) v = fmaxf(v, __shfl_xor(v, m, 32));
  return v;
}
__device__ __forceinline__ float red_sum16(float v) {
#pragma unroll
  for (int m = 1; m < 16; m <<= 1) v += __shfl_xor(v, m, 32);
  return v;
}

// ---------------- per-mode key/value row addressing ----------------
// MODE 0: keys = k_rep rows, values = v_mean rows            (part 0)
// MODE 1: keys = k_rep[sampled idx], values = v[sampled abs] (part 1, "del")
// MODE 2: keys = k[sampled abs],     values = v[sampled abs] (part 1, "add")
// MODE 3: keys = k rows,             values = v rows          (part 2)
template <int MODE>
__device__ __forceinline__ void row_ptrs(int kg, int bh,
                                         const float* __restrict__ kglob,
                                         const float* __restrict__ vglob,
                                         const float* __restrict__ krep,
                                         const float* __restrict__ vmean,
                                         const int* __restrict__ kidx,
                                         const float** kp, const float** vp) {
  if (MODE == 0) {
    const size_t off = ((size_t)bh * PLEN + kg) * DIM;
    *kp = krep + off;
    *vp = vmean + off;
  } else if (MODE == 3) {
    const size_t off = ((size_t)bh * NSEQ + kg) * DIM;
    *kp = kglob + off;
    *vp = vglob + off;
  } else {
    const int g = kg / MSMP;
    const int ii = kg - g * MSMP;
    const int wi = kidx[(bh * GRP + g) * MSMP + ii];
    const size_t voff = ((size_t)bh * NSEQ + (size_t)g * PLEN + wi) * DIM;
    *vp = vglob + voff;
    *kp = (MODE == 1) ? (krep + ((size_t)bh * PLEN + wi) * DIM) : (kglob + voff);
  }
}

// Issue one 32-key chunk of K and V (f32) into LDS via async loads.
// 512 16B segments per matrix, 2 per thread -> 4 async instructions per wave.
template <int MODE>
__device__ __forceinline__ void issue_chunk(int kb, int nkeys, int bh,
                                            const float* __restrict__ kglob,
                                            const float* __restrict__ vglob,
                                            const float* __restrict__ krep,
                                            const float* __restrict__ vmean,
                                            const int* __restrict__ kidx,
                                            float* bufK, float* bufV) {
#if HAS_ASYNC
  const int tid = threadIdx.x;
#pragma unroll
  for (int it = 0; it < 2; ++it) {
    const int e = tid + it * 256;  // 0..511
    const int j = e >> 4;          // key row within tile
    const int s = e & 15;          // 16-byte segment within the 256B row
    int kg = kb + j;
    if (kg >= nkeys) kg = kb;      // clamp; S-mask / P==0 neutralize these later
    const float* kp;
    const float* vp;
    row_ptrs<MODE>(kg, bh, kglob, vglob, krep, vmean, kidx, &kp, &vp);
    __builtin_amdgcn_global_load_async_to_lds_b128(
        (gv4i*)(kp + s * 4), (lv4i*)(bufK + j * DIM + s * 4), 0, 0);
    __builtin_amdgcn_global_load_async_to_lds_b128(
        (gv4i*)(vp + s * 4), (lv4i*)(bufV + j * DIM + s * 4), 0, 0);
  }
#endif
}

// ---------------- flash attention pass over a 16-row Q tile ----------------
template <int MODE>
__device__ __forceinline__ void flash_pass(
    int nkeys, int bh,
    const float* __restrict__ kglob, const float* __restrict__ vglob,
    const float* __restrict__ krep,  const float* __restrict__ vmean,
    const int* __restrict__ kidx,
    const bf16* qt, bf16* KT, bf16* VT, bf16* Pt,
    float* bK0, float* bK1, float* bV0, float* bV1,
    int lane, float* m, float* l, float o[4][8]) {
  const int tid = threadIdx.x;
  const int half8 = (lane & 16) ? 8 : 0;
  const int nrow = lane & 15;
  const int chunks = (nkeys + 31) >> 5;
#if HAS_ASYNC
  issue_chunk<MODE>(0, nkeys, bh, kglob, vglob, krep, vmean, kidx, bK0, bV0);
#endif
  for (int chk = 0; chk < chunks; ++chk) {
    const int kb = chk << 5;
#if HAS_ASYNC
    // prefetch chunk k+1 into the alternate buffer, wait for chunk k only
    if (chk + 1 < chunks) {
      issue_chunk<MODE>(kb + 32, nkeys, bh, kglob, vglob, krep, vmean, kidx,
                        ((chk + 1) & 1) ? bK1 : bK0, ((chk + 1) & 1) ? bV1 : bV0);
      __builtin_amdgcn_s_wait_asynccnt(4);
    } else {
      __builtin_amdgcn_s_wait_asynccnt(0);
    }
    __syncthreads();
    {  // LDS f32 -> bf16 tiles: KT row-major [32][64], VT transposed [64][32]
      const float* bK = (chk & 1) ? bK1 : bK0;
      const float* bV = (chk & 1) ? bV1 : bV0;
      for (int e = tid; e < 32 * DIM; e += 256) {
        const int j = e >> 6, c = e & (DIM - 1);
        KT[j * KT_S + c] = (bf16)bK[e];
        VT[c * VT_S + j] = (bf16)bV[e];
      }
    }
#else
    for (int e = tid; e < 32 * DIM; e += 256) {
      const int j = e >> 6, c = e & (DIM - 1);
      int kg = kb + j;
      if (kg >= nkeys) kg = kb;
      const float* kp;
      const float* vp;
      row_ptrs<MODE>(kg, bh, kglob, vglob, krep, vmean, kidx, &kp, &vp);
      if (MODE == 3 && ((e & 63) == 0) && (kb + 32 < nkeys))
        __builtin_prefetch(kp + 32 * DIM, 0, 1);
      KT[j * KT_S + c] = (bf16)kp[c];
      VT[c * VT_S + j] = (bf16)vp[c];
    }
#endif
    __syncthreads();

    // S = (Q K^T) * d^-0.5, two 16x16 tiles (keys kb..kb+15, kb+16..kb+31)
    float s0[8], s1[8];
    {
      v8f a0 = {}, a1 = {};
#pragma unroll
      for (int kk = 0; kk < DIM; kk += 32) {
        bf16x16 fa = load_fragA(qt, nrow, DIM, kk, lane);
        bf16x16 fb0 = load_fragB(KT, nrow, KT_S, kk, lane);
        bf16x16 fb1 = load_fragB(KT + 16 * KT_S, nrow, KT_S, kk, lane);
        a0 = __builtin_amdgcn_wmma_f32_16x16x32_bf16(false, fa, false, fb0,
                                                     (short)0, a0, false, false);
        a1 = __builtin_amdgcn_wmma_f32_16x16x32_bf16(false, fa, false, fb1,
                                                     (short)0, a1, false, false);
      }
      const bool inv0 = (kb + nrow) >= nkeys;
      const bool inv1 = (kb + 16 + nrow) >= nkeys;
#pragma unroll
      for (int r = 0; r < 8; ++r) {
        s0[r] = inv0 ? NEG_BIG : a0[r] * 0.125f;  // d^-0.5 = 1/8
        s1[r] = inv1 ? NEG_BIG : a1[r] * 0.125f;
      }
    }
    // online softmax update; write P to LDS in row-major 16x32 (A-layout source)
#pragma unroll
    for (int r = 0; r < 8; ++r) {
      const float rm = red_max16(fmaxf(s0[r], s1[r]));
      const float mnew = fmaxf(m[r], rm);
      const float p0 = __expf(s0[r] - mnew);
      const float p1 = __expf(s1[r] - mnew);
      const float rowsum = red_sum16(p0 + p1);
      const float sc = __expf(m[r] - mnew);
      l[r] = l[r] * sc + rowsum;
      m[r] = mnew;
#pragma unroll
      for (int t = 0; t < 4; ++t) o[t][r] *= sc;
      const int M = r + half8;
      Pt[M * PT_S + nrow] = (bf16)p0;
      Pt[M * PT_S + 16 + nrow] = (bf16)p1;
    }
    // O += P V   (A = P 16x32, B = V 32x16 per output-column tile, from V^T rows)
    {
      bf16x16 fa = load_fragA(Pt, nrow, PT_S, 0, lane);
#pragma unroll
      for (int t = 0; t < 4; ++t) {
        bf16x16 fb = load_fragB(VT, t * 16 + nrow, VT_S, 0, lane);
        v8f acc;
#pragma unroll
        for (int i = 0; i < 8; ++i) acc[i] = o[t][i];
        acc = __builtin_amdgcn_wmma_f32_16x16x32_bf16(false, fa, false, fb,
                                                      (short)0, acc, false, false);
#pragma unroll
        for (int i = 0; i < 8; ++i) o[t][i] = acc[i];
      }
    }
    __syncthreads();
  }
}

// ---------------- v_mean = mean over groups ----------------
__global__ __launch_bounds__(256) void vmean_kernel(const float* __restrict__ v,
                                                    float* __restrict__ vmean) {
  const size_t idx = (size_t)blockIdx.x * 256 + threadIdx.x;
  if (idx >= (size_t)BH * PLEN * DIM) return;
  const size_t bh = idx / ((size_t)PLEN * DIM);
  const size_t rem = idx - bh * (size_t)PLEN * DIM;
  float acc = 0.f;
#pragma unroll
  for (int g = 0; g < GRP; ++g)
    acc += v[bh * (size_t)NSEQ * DIM + (size_t)g * PLEN * DIM + rem];
  vmean[idx] = acc * 0.125f;
}

// ---------------- significant-channel median representative ----------------
__global__ __launch_bounds__(256) void median_rep_kernel(const float* __restrict__ src,
                                                         float* __restrict__ rep) {
  const int wv = blockIdx.x * 8 + (threadIdx.x >> 5);
  const int lane = threadIdx.x & 31;
  const int bh = wv >> 10;
  const int n = wv & (PLEN - 1);
  const float* s = src + (size_t)bh * NSEQ * DIM;
  float mv0 = 0.f, mv1 = 0.f;
#pragma unroll
  for (int g = 0; g < GRP; ++g) {
    const size_t row = ((size_t)g * PLEN + n) * DIM;
    mv0 += s[row + lane];
    mv1 += s[row + lane + 32];
  }
  mv0 *= 0.125f;
  mv1 *= 0.125f;
  int ch[3];
#pragma unroll
  for (int t = 0; t < 3; ++t) {  // top-3 channels of the group mean
    float bv;
    int bi;
    if (mv0 >= mv1) { bv = mv0; bi = lane; } else { bv = mv1; bi = lane + 32; }
#pragma unroll
    for (int mk = 16; mk >= 1; mk >>= 1) {
      const float ov = __shfl_xor(bv, mk, 32);
      const int oi = __shfl_xor(bi, mk, 32);
      if (ov > bv || (ov == bv && oi < bi)) { bv = ov; bi = oi; }
    }
    ch[t] = bi;
    if (bi == lane) mv0 = NEG_BIG;
    if (bi == lane + 32) mv1 = NEG_BIG;
  }
  float gval = 0.f;
  if (lane < GRP) {
    const size_t row = ((size_t)lane * PLEN + n) * DIM;
    gval = s[row + ch[0]] * 100.f + s[row + ch[1]] * 10.f + s[row + ch[2]];
  }
  float vals[GRP];
#pragma unroll
  for (int g = 0; g < GRP; ++g) vals[g] = __shfl(gval, g, 32);
  int med = 0;
#pragma unroll
  for (int g = 0; g < GRP; ++g) {  // stable ascending rank; take rank (G-1)/2
    int rank = 0;
#pragma unroll
    for (int g2 = 0; g2 < GRP; ++g2)
      rank += (vals[g2] < vals[g]) || (vals[g2] == vals[g] && g2 < g);
    if (rank == (GRP - 1) / 2) med = g;
  }
  const size_t srow = ((size_t)med * PLEN + n) * DIM;
  float* r = rep + ((size_t)bh * PLEN + n) * DIM;
  r[lane] = s[srow + lane];
  r[lane + 32] = s[srow + lane + 32];
}

// ---------------- norm-guided Gumbel top-k sampling ----------------
__device__ __forceinline__ unsigned hash_u32(unsigned x) {
  x ^= x >> 17; x *= 0xed5ad4bbu;
  x ^= x >> 11; x *= 0xac4c1b51u;
  x ^= x >> 15; x *= 0x31848babu;
  x ^= x >> 14;
  return x;
}

__global__ __launch_bounds__(256) void sample_kernel(const float* __restrict__ src,
                                                     const float* __restrict__ rep,
                                                     int* __restrict__ idx_out,
                                                     unsigned seed) {
  __shared__ float score[PLEN];
  __shared__ float rv[256];
  __shared__ int ri[256];
  const int tid = threadIdx.x;
  const int bh = blockIdx.x >> 3;
  const int g = blockIdx.x & 7;
  for (int i = tid; i < PLEN; i += 256) {
    const float* s = src + ((size_t)bh * NSEQ + (size_t)g * PLEN + i) * DIM;
    const float* r = rep + ((size_t)bh * PLEN + i) * DIM;
    float acc = 0.f;
#pragma unroll 8
    for (int c = 0; c < DIM; ++c) {
      const float d = s[c] - r[c];
      acc += d * d;
    }
    const float prob = sqrtf(acc) + EPS_F;
    const unsigned h = hash_u32(seed * 0x9E3779B9u + (unsigned)(blockIdx.x * PLEN + i));
    const float u = ((float)h + 0.5f) * 2.3283064365386963e-10f;
    score[i] = __logf(prob) - __logf(-__logf(u));
  }
  __syncthreads();
  for (int t = 0; t < MSMP; ++t) {
    float bv = NEG_BIG;
    int bi = 0;
    for (int i = tid; i < PLEN; i += 256)
      if (score[i] > bv || (score[i] == bv && i < bi)) { bv = score[i]; bi = i; }
    rv[tid] = bv;
    ri[tid] = bi;
    __syncthreads();
    for (int s2 = 128; s2 > 0; s2 >>= 1) {
      if (tid < s2) {
        if (rv[tid + s2] > rv[tid] ||
            (rv[tid + s2] == rv[tid] && ri[tid + s2] < ri[tid])) {
          rv[tid] = rv[tid + s2];
          ri[tid] = ri[tid + s2];
        }
      }
      __syncthreads();
    }
    if (tid == 0) {
      idx_out[blockIdx.x * MSMP + t] = ri[0];
      score[ri[0]] = NEG_BIG;
    }
    __syncthreads();
  }
}

#if HAS_ASYNC
#define DECL_ASYNC_BUFS                                \
  __shared__ __align__(16) float sKF[2][32 * DIM];     \
  __shared__ __align__(16) float sVF[2][32 * DIM];
#define ASYNC_BUF_ARGS sKF[0], sKF[1], sVF[0], sVF[1]
#else
#define DECL_ASYNC_BUFS
#define ASYNC_BUF_ARGS nullptr, nullptr, nullptr, nullptr
#endif

// ---------------- part 0 + part 1 fused ----------------
__global__ __launch_bounds__(256) void part01_kernel(
    const float* __restrict__ k, const float* __restrict__ v,
    const float* __restrict__ qrep, const float* __restrict__ krep,
    const float* __restrict__ vmean, const int* __restrict__ kidx,
    float* __restrict__ out_attn, float* __restrict__ out_lse) {
  __shared__ __align__(16) bf16 sKT[32 * KT_S];
  __shared__ __align__(16) bf16 sVT[DIM * VT_S];
  __shared__ __align__(16) bf16 sQT[8 * 16 * DIM];
  __shared__ __align__(16) bf16 sP[8 * 16 * PT_S];
  DECL_ASYNC_BUFS
  const int tid = threadIdx.x, wave = tid >> 5, lane = tid & 31;
  const int bh = blockIdx.y;
  const int qbase = blockIdx.x * 128 + wave * 16;
  bf16* qt = sQT + wave * 16 * DIM;
  bf16* Pt = sP + wave * 16 * PT_S;
  for (int e = lane; e < 16 * DIM; e += 32) {
    const int rr = e >> 6, c = e & (DIM - 1);
    qt[rr * DIM + c] = (bf16)qrep[((size_t)bh * PLEN + qbase + rr) * DIM + c];
  }
  const int half8 = (lane & 16) ? 8 : 0;
  const int nrow = lane & 15;

  float m0[8], l0[8], o0[4][8];
#pragma unroll
  for (int r = 0; r < 8; ++r) {
    m0[r] = NEG_BIG; l0[r] = 0.f;
#pragma unroll
    for (int t = 0; t < 4; ++t) o0[t][r] = 0.f;
  }
  // part 0: q_rep vs k_rep, values = v_mean; lse += ln(gk)
  flash_pass<0>(PLEN, bh, k, v, krep, vmean, kidx, qt, sKT, sVT, Pt,
                ASYNC_BUF_ARGS, lane, m0, l0, o0);
  float a[4][8], lse[8];
#pragma unroll
  for (int r = 0; r < 8; ++r) {
    const float il = 1.f / l0[r];
    lse[r] = m0[r] + __logf(l0[r]) + LOG_G;
#pragma unroll
    for (int t = 0; t < 4; ++t) a[t][r] = o0[t][r] * il;
  }
  // part 1 "del": q_rep vs k_rep_sub
#pragma unroll
  for (int r = 0; r < 8; ++r) {
    m0[r] = NEG_BIG; l0[r] = 0.f;
#pragma unroll
    for (int t = 0; t < 4; ++t) o0[t][r] = 0.f;
  }
  flash_pass<1>(NSUB, bh, k, v, krep, vmean, kidx, qt, sKT, sVT, Pt,
                ASYNC_BUF_ARGS, lane, m0, l0, o0);
#pragma unroll
  for (int r = 0; r < 8; ++r) {  // _sub_lse
    const float ldel = m0[r] + __logf(l0[r]);
    const float il = 1.f / l0[r];
    const float rr_ = __expf(ldel - lse[r]);
    const float c_ = 1.f / (1.f - rr_);
#pragma unroll
    for (int t = 0; t < 4; ++t) a[t][r] = c_ * a[t][r] + (1.f - c_) * (o0[t][r] * il);
    lse[r] = lse[r] + __logf(fmaxf(1.f - rr_, EPS_F));
  }
  // part 1 "add": q_rep vs k_sub
#pragma unroll
  for (int r = 0; r < 8; ++r) {
    m0[r] = NEG_BIG; l0[r] = 0.f;
#pragma unroll
    for (int t = 0; t < 4; ++t) o0[t][r] = 0.f;
  }
  flash_pass<2>(NSUB, bh, k, v, krep, vmean, kidx, qt, sKT, sVT, Pt,
                ASYNC_BUF_ARGS, lane, m0, l0, o0);
#pragma unroll
  for (int r = 0; r < 8; ++r) {  // _add_lse
    const float ladd = m0[r] + __logf(l0[r]);
    const float il = 1.f / l0[r];
    const float c2 = 1.f / (1.f + __expf(ladd - lse[r]));
#pragma unroll
    for (int t = 0; t < 4; ++t) a[t][r] = c2 * a[t][r] + (1.f - c2) * (o0[t][r] * il);
    lse[r] = lse[r] - __logf(c2 + EPS_F);
  }
  // broadcast to all gq group copies
#pragma unroll
  for (int r = 0; r < 8; ++r) {
    const int row = qbase + r + half8;
#pragma unroll
    for (int g = 0; g < GRP; ++g) {
      const size_t base = ((size_t)bh * NSEQ + (size_t)g * PLEN + row) * DIM;
#pragma unroll
      for (int t = 0; t < 4; ++t) out_attn[base + t * 16 + nrow] = a[t][r];
    }
    if (nrow == 0) {
#pragma unroll
      for (int g = 0; g < GRP; ++g)
        out_lse[(size_t)bh * NSEQ + (size_t)g * PLEN + row] = lse[r];
    }
  }
}

// ---------------- part 2: exact rows for sampled queries ----------------
__global__ __launch_bounds__(256) void part2_kernel(
    const float* __restrict__ q, const float* __restrict__ k,
    const float* __restrict__ v, const int* __restrict__ qidx,
    float* __restrict__ out_attn, float* __restrict__ out_lse) {
  __shared__ __align__(16) bf16 sKT[32 * KT_S];
  __shared__ __align__(16) bf16 sVT[DIM * VT_S];
  __shared__ __align__(16) bf16 sQT[8 * 16 * DIM];
  __shared__ __align__(16) bf16 sP[8 * 16 * PT_S];
  DECL_ASYNC_BUFS
  const int tid = threadIdx.x, wave = tid >> 5, lane = tid & 31;
  const int bh = blockIdx.y;
  const int qtb = blockIdx.x * 128 + wave * 16;  // 0..383, rows >= 272 masked
  bf16* qt = sQT + wave * 16 * DIM;
  bf16* Pt = sP + wave * 16 * PT_S;
  for (int e = lane; e < 16 * DIM; e += 32) {
    const int rr = e >> 6, c = e & (DIM - 1);
    const int sr = qtb + rr;
    float val = 0.f;
    if (sr < NSUB) {
      const int g = sr / MSMP, ii = sr - (sr / MSMP) * MSMP;
      const int wi = qidx[(bh * GRP + g) * MSMP + ii];
      val = q[((size_t)bh * NSEQ + (size_t)g * PLEN + wi) * DIM + c];
    }
    qt[rr * DIM + c] = (bf16)val;
  }
  const int half8 = (lane & 16) ? 8 : 0;
  const int nrow = lane & 15;
  float m0[8], l0[8], o0[4][8];
#pragma unroll
  for (int r = 0; r < 8; ++r) {
    m0[r] = NEG_BIG; l0[r] = 0.f;
#pragma unroll
    for (int t = 0; t < 4; ++t) o0[t][r] = 0.f;
  }
  flash_pass<3>(NSEQ, bh, k, v, nullptr, nullptr, nullptr, qt, sKT, sVT, Pt,
                ASYNC_BUF_ARGS, lane, m0, l0, o0);
  // scatter into broadcast output (runs after part01 on the stream)
#pragma unroll
  for (int r = 0; r < 8; ++r) {
    const int sr = qtb + r + half8;
    if (sr < NSUB) {
      const int g = sr / MSMP, ii = sr - (sr / MSMP) * MSMP;
      const int wi = qidx[(bh * GRP + g) * MSMP + ii];
      const size_t base = ((size_t)bh * NSEQ + (size_t)g * PLEN + wi) * DIM;
      const float il = 1.f / l0[r];
#pragma unroll
      for (int t = 0; t < 4; ++t) out_attn[base + t * 16 + nrow] = o0[t][r] * il;
      if (nrow == 0)
        out_lse[(size_t)bh * NSEQ + (size_t)g * PLEN + wi] = m0[r] + __logf(l0[r]);
    }
  }
}

extern "C" void kernel_launch(void* const* d_in, const int* in_sizes, int n_in,
                              void* d_out, int out_size, void* d_ws, size_t ws_size,
                              hipStream_t stream) {
  const float* q = (const float*)d_in[0];
  const float* k = (const float*)d_in[1];
  const float* v = (const float*)d_in[2];
  // d_in[3]/d_in[4]: n_query_groups / n_key_groups (fixed = 8 for this problem)

  float* qrep = (float*)d_ws;
  float* krep = qrep + (size_t)BH * PLEN * DIM;
  float* vmean = krep + (size_t)BH * PLEN * DIM;
  int* kidx = (int*)(vmean + (size_t)BH * PLEN * DIM);
  int* qidx = kidx + BH * GRP * MSMP;

  float* out_attn = (float*)d_out;
  float* out_lse = out_attn + (size_t)BH * NSEQ * DIM;

  vmean_kernel<<<(BH * PLEN * DIM) / 256, 256, 0, stream>>>(v, vmean);
  median_rep_kernel<<<(BH * PLEN) / 8, 256, 0, stream>>>(q, qrep);
  median_rep_kernel<<<(BH * PLEN) / 8, 256, 0, stream>>>(k, krep);
  sample_kernel<<<BH * GRP, 256, 0, stream>>>(k, krep, kidx, 1u);
  sample_kernel<<<BH * GRP, 256, 0, stream>>>(q, qrep, qidx, 2u);
  part01_kernel<<<dim3(PLEN / 128, BH), 256, 0, stream>>>(k, v, qrep, krep, vmean,
                                                          kidx, out_attn, out_lse);
  part2_kernel<<<dim3(3, BH), 256, 0, stream>>>(q, k, v, qidx, out_attn, out_lse);
}
